// HierarchicalRouter_46084999086157
// MI455X (gfx1250) — matrix operand order repaired and checked by reading, not compiled, not run on hardware
//
#include <hip/hip_runtime.h>
#include <hip/hip_bf16.h>
#include <math.h>

// ---------------------------------------------------------------------------
// Hierarchical MoE router for MI455X (gfx1250).
//   logits = x[16384x2048] @ concat(group_gate, expert_gate)^T   (72 cols)
//   then per-token 2-level softmax / threshold / renormalize.
// GEMM on the bf16 WMMA pipe with an fp32-accurate bf16 hi/lo split
// (3 x v_wmma_f32_16x16x32_bf16 per K=32). Split-K=2 per block doubles the
// wave count (2048 waves) so memory latency hides behind XDL work.
// ---------------------------------------------------------------------------

#define N_TOK   16384
#define N_EMBD  2048
#define C_PAD   80          // 72 real cols padded to 5 tiles of 16
#define NTILES  5
#define MT_PER_BLOCK 4      // token tiles (16 tokens each) per block
#define KSPLIT  2           // K-halves per token tile
#define WPB     (MT_PER_BLOCK * KSPLIT)   // 8 waves per block
#define TPB     (MT_PER_BLOCK * 16)       // 64 tokens per block
#define KHALF_LEN (N_EMBD / KSPLIT)       // 1024

typedef __attribute__((ext_vector_type(16))) __bf16 v16bf;
typedef __attribute__((ext_vector_type(8)))  float  v8f;

union AFrag {
    v16bf v;
    unsigned int u[8];
};
union BFrag {
    v16bf v;
    uint4 q[2];
};

__device__ __forceinline__ unsigned int rnd_bf(float f) {
    unsigned int u = __float_as_uint(f);
    return u + 0x7FFFu + ((u >> 16) & 1u);   // RNE in the high 16 bits
}
__device__ __forceinline__ unsigned short f2bf_rne(float f) {
    return (unsigned short)(rnd_bf(f) >> 16);
}
__device__ __forceinline__ float bf2f(unsigned short s) {
    return __uint_as_float(((unsigned int)s) << 16);
}

// ---------------------------------------------------------------------------
// Prep: split fp32 gate weights into bf16 hi/lo planes, padded to 80 rows.
// w_hi/w_lo are [C_PAD][N_EMBD] row-major bf16 (as ushort).
// ---------------------------------------------------------------------------
__global__ void router_prep_w(const float* __restrict__ gw,   // [8][2048]
                              const float* __restrict__ ew,   // [64][2048]
                              unsigned short* __restrict__ w_hi,
                              unsigned short* __restrict__ w_lo) {
    int idx = blockIdx.x * blockDim.x + threadIdx.x;
    if (idx >= C_PAD * N_EMBD) return;
    int c = idx / N_EMBD;
    int k = idx - c * N_EMBD;
    float v = 0.0f;
    if (c < 8)        v = gw[c * N_EMBD + k];
    else if (c < 72)  v = ew[(c - 8) * N_EMBD + k];
    unsigned short hi = f2bf_rne(v);
    unsigned short lo = f2bf_rne(v - bf2f(hi));
    w_hi[idx] = hi;
    w_lo[idx] = lo;
}

// ---------------------------------------------------------------------------
// Main: split-K bf16x3 WMMA GEMM + per-token hierarchical softmax epilogue.
// ---------------------------------------------------------------------------
__global__ __launch_bounds__(WPB * 32)
void router_main(const float* __restrict__ x,
                 const unsigned short* __restrict__ w_hi,
                 const unsigned short* __restrict__ w_lo,
                 float* __restrict__ out_mask,   // [N_TOK][64] (0.0 / 1.0)
                 float* __restrict__ out_wts) {  // [N_TOK][64]
    __shared__ float logits[WPB][16][C_PAD];     // 40 KB

    const int tid   = threadIdx.x;
    const int wave  = tid >> 5;
    const int lane  = tid & 31;
    const int l16   = lane & 15;
    const int khalf = lane >> 4;        // which K half-group this lane holds
    const int mtile = wave & (MT_PER_BLOCK - 1);
    const int kpart = wave >> 2;        // 0 or 1: which 1024-wide K slab

    const int tokBase = (blockIdx.x * MT_PER_BLOCK + mtile) * 16;
    const int kbegin  = kpart * KHALF_LEN;

    v8f acc[NTILES];
    const v8f vzero = {0.f, 0.f, 0.f, 0.f, 0.f, 0.f, 0.f, 0.f};
#pragma unroll
    for (int t = 0; t < NTILES; ++t) acc[t] = vzero;

    // A layout (16-bit A 16x32, contiguous K order used here): lane m = l16,
    //  lanes 0-15:  halves 0..15 = K 0..15
    //  lanes 16-31: halves 0..15 = K 16..31
    const float* arow = x + (size_t)(tokBase + l16) * N_EMBD;

#pragma unroll 1
    for (int k0 = kbegin; k0 < kbegin + KHALF_LEN; k0 += 32) {
        if (k0 + 32 < kbegin + KHALF_LEN)
            __builtin_prefetch(arow + k0 + 32, 0, 1);

        const float* p = arow + k0 + khalf * 16;
        float4 u0 = *reinterpret_cast<const float4*>(p);
        float4 u1 = *reinterpret_cast<const float4*>(p + 4);
        float4 u2 = *reinterpret_cast<const float4*>(p + 8);
        float4 u3 = *reinterpret_cast<const float4*>(p + 12);
        float af[16] = {u0.x, u0.y, u0.z, u0.w, u1.x, u1.y, u1.z, u1.w,
                        u2.x, u2.y, u2.z, u2.w, u3.x, u3.y, u3.z, u3.w};

        // hi = RNE bf16 of af; lo = RNE bf16 of (af - hi). Pack pairs with
        // v_perm_b32: result = { hi(odd)[31:16], hi(even)[31:16] }.
        AFrag ahi, alo;
        unsigned int r[16];
        float lo[16];
#pragma unroll
        for (int i = 0; i < 16; ++i) {
            r[i]  = rnd_bf(af[i]);
            lo[i] = af[i] - __uint_as_float(r[i] & 0xFFFF0000u);
        }
#pragma unroll
        for (int j = 0; j < 8; ++j)
            ahi.u[j] = __builtin_amdgcn_perm(r[2 * j + 1], r[2 * j], 0x07060302u);
#pragma unroll
        for (int i = 0; i < 16; ++i) r[i] = rnd_bf(lo[i]);
#pragma unroll
        for (int j = 0; j < 8; ++j)
            alo.u[j] = __builtin_amdgcn_perm(r[2 * j + 1], r[2 * j], 0x07060302u);

        // B layout (32x16, lane = N): lanes 0-15 hold K k0..k0+15,
        // lanes 16-31 hold K k0+16..k0+31 (16 contiguous bf16 per lane).
#pragma unroll
        for (int nt = 0; nt < NTILES; ++nt) {
            size_t boff = (size_t)(nt * 16 + l16) * N_EMBD + k0 + khalf * 16;
            BFrag bhi, blo;
            bhi.q[0] = *reinterpret_cast<const uint4*>(w_hi + boff);
            bhi.q[1] = *reinterpret_cast<const uint4*>(w_hi + boff + 8);
            blo.q[0] = *reinterpret_cast<const uint4*>(w_lo + boff);
            blo.q[1] = *reinterpret_cast<const uint4*>(w_lo + boff + 8);

            acc[nt] = __builtin_amdgcn_wmma_f32_16x16x32_bf16(
                false, ahi.v, false, bhi.v, (short)0, acc[nt], false, false);
            acc[nt] = __builtin_amdgcn_wmma_f32_16x16x32_bf16(
                false, ahi.v, false, blo.v, (short)0, acc[nt], false, false);
            acc[nt] = __builtin_amdgcn_wmma_f32_16x16x32_bf16(
                false, alo.v, false, bhi.v, (short)0, acc[nt], false, false);
        }
    }

    // C/D layout: VGPR r, lanes 0-15 -> (M=r, N=lane); lanes 16-31 -> (M=8+r).
#pragma unroll
    for (int nt = 0; nt < NTILES; ++nt)
#pragma unroll
        for (int r8 = 0; r8 < 8; ++r8)
            logits[wave][khalf * 8 + r8][nt * 16 + l16] = acc[nt][r8];

    __syncthreads();

    // ---------------- per-token epilogue ----------------------------------
    // Waves 0..3 handle their token tile; lanes 0..15 take one token each.
    // Combined logit c = slab(kpart=0)[c] + slab(kpart=1)[c].
    if (wave < MT_PER_BLOCK && lane < 16) {
        float* L0 = logits[wave][lane];                    // K-half 0 (also stash)
        const float* L1 = logits[wave + MT_PER_BLOCK][lane]; // K-half 1

        // level 1: group softmax over cols 0..7
        float gl[8];
#pragma unroll
        for (int g = 0; g < 8; ++g) gl[g] = L0[g] + L1[g];
        float gmax = -INFINITY;
#pragma unroll
        for (int g = 0; g < 8; ++g) gmax = fmaxf(gmax, gl[g]);
        float ge[8];
        float gsum = 0.0f;
#pragma unroll
        for (int g = 0; g < 8; ++g) { ge[g] = expf(gl[g] - gmax); gsum += ge[g]; }
        const float ginv = 1.0f / gsum;

        // level 2: expert softmax per group; stash selected weights into
        // L0[0..63] (writes land only on already-consumed columns).
        float wsum = 0.0f;
#pragma unroll 1
        for (int g = 0; g < 8; ++g) {
            const float gp  = ge[g] * ginv;
            const bool  gok = (gp >= 0.125f);
            float el[8];
#pragma unroll
            for (int e = 0; e < 8; ++e) el[e] = L0[8 + g * 8 + e] + L1[8 + g * 8 + e];
            float emax = -INFINITY;
#pragma unroll
            for (int e = 0; e < 8; ++e) emax = fmaxf(emax, el[e]);
            float ep[8];
            float esum = 0.0f;
#pragma unroll
            for (int e = 0; e < 8; ++e) { ep[e] = expf(el[e] - emax); esum += ep[e]; }
            const float einv = 1.0f / esum;
#pragma unroll
            for (int e = 0; e < 8; ++e) {
                float pe = ep[e] * einv;
                bool  vm = gok && (pe >= 0.125f);
                float w  = vm ? gp * pe : 0.0f;   // valid => w >= 1/64 > 0
                L0[g * 8 + e] = w;
                wsum += w;
            }
        }
        const float inv = 1.0f / fmaxf(wsum, 1e-9f);

        const int token = tokBase + lane;
        float* om = out_mask + (size_t)token * 64;
        float* ow = out_wts  + (size_t)token * 64;
#pragma unroll
        for (int i = 0; i < 64; ++i) {
            float w = L0[i];
            om[i] = (w > 0.0f) ? 1.0f : 0.0f;
            ow[i] = w * inv;
        }
    }
}

// ---------------------------------------------------------------------------
extern "C" void kernel_launch(void* const* d_in, const int* in_sizes, int n_in,
                              void* d_out, int out_size, void* d_ws, size_t ws_size,
                              hipStream_t stream) {
    const float* x  = (const float*)d_in[0];   // [16384, 2048] f32
    const float* gw = (const float*)d_in[1];   // [8, 2048]     f32
    const float* ew = (const float*)d_in[2];   // [64, 2048]    f32

    // workspace: bf16 hi/lo planes of padded weights, 2 * 80*2048*2B = 640 KB
    unsigned short* w_hi = (unsigned short*)d_ws;
    unsigned short* w_lo = w_hi + (size_t)C_PAD * N_EMBD;

    float* out_mask = (float*)d_out;                       // [16384, 64]
    float* out_wts  = out_mask + (size_t)N_TOK * 64;       // [16384, 64]

    const int prepN = C_PAD * N_EMBD;
    router_prep_w<<<(prepN + 255) / 256, 256, 0, stream>>>(gw, ew, w_hi, w_lo);

    const int blocks = N_TOK / TPB;                        // 256 blocks, 8 waves
    router_main<<<blocks, WPB * 32, 0, stream>>>(x, w_hi, w_lo, out_mask, out_wts);
}
